// mLSTMWeavingBackend_42159398978079
// MI455X (gfx1250) — matrix operand order repaired and checked by reading, not compiled, not run on hardware
//
#include <hip/hip_runtime.h>
#include <math.h>

#define S_LEN   2048
#define DHD     64
#define BH      64        // B*NH
#define CHUNK   32
#define NCHUNK  (S_LEN / CHUNK)
#define SLABS   4
#define VSLAB   16
#define EPSV    1e-6f
#define INV_SQRT_DH 0.125f

typedef __attribute__((ext_vector_type(16))) __bf16 v16bf;
typedef __attribute__((ext_vector_type(8)))  float  v8f;

__global__ __launch_bounds__(32)
void mlstm_chunk_wmma(const float* __restrict__ q,  const float* __restrict__ k,
                      const float* __restrict__ v,  const float* __restrict__ ivec,
                      const float* __restrict__ fvec,
                      const float* __restrict__ c0g, const float* __restrict__ n0g,
                      const float* __restrict__ m0g, float* __restrict__ dout)
{
  const int lane = threadIdx.x;
  const int bh   = blockIdx.x >> 2;
  const int slab = blockIdx.x & 3;
  const int c0   = slab * VSLAB;
  const int hl   = lane >> 4;       // half-wave
  const int l15  = lane & 15;

  // LDS: padded strides for bank spread; 8B-aligned rows for float2 stores
  __shared__ float   Qc[CHUNK][66];     // f32 (also used for qn0 dot)
  __shared__ __bf16  Kb[CHUNK][66];     // bf16: converted once at staging
  __shared__ float   Vc[CHUNK][18];
  __shared__ __bf16  Sl[CHUNK][34];     // bf16 S tiles (A-layout source)
  __shared__ __bf16  Cb[DHD][17];       // bf16 shadow of C state
  __shared__ float   n_lds[DHD];
  __shared__ float rowfac_l[CHUNK], colfac_l[CHUNK], w_l[CHUNK], g_l[CHUNK], rden_l[CHUNK];

  const float* qg  = q    + (size_t)bh * S_LEN * DHD;
  const float* kg  = k    + (size_t)bh * S_LEN * DHD;
  const float* vg  = v    + (size_t)bh * S_LEN * DHD;
  const float* igp = ivec + (size_t)bh * S_LEN;
  const float* fgp = fvec + (size_t)bh * S_LEN;
  float*       og  = dout + (size_t)bh * S_LEN * DHD;

  // ---- load initial state: C slab (f32 regs, WMMA C/D layout) + bf16 shadow ----
  v8f accC[4];
  #pragma unroll
  for (int rt = 0; rt < 4; ++rt)
    #pragma unroll
    for (int r = 0; r < 8; ++r) {
      int dk = rt * 16 + r + hl * 8;
      float cv = c0g[(size_t)bh * DHD * DHD + (size_t)dk * DHD + c0 + l15];
      accC[rt][r] = cv;
      Cb[dk][l15] = (__bf16)cv;
    }
  n_lds[lane]      = n0g[bh * DHD + lane];
  n_lds[lane + 32] = n0g[bh * DHD + lane + 32];
  float m_prev = m0g[bh];

  // tile (0,1) of S is fully causal-masked and never overwritten: zero once
  #pragma unroll
  for (int r = 0; r < 8; ++r) Sl[r + hl * 8][16 + l15] = (__bf16)0.f;

  for (int ch = 0; ch < NCHUNK; ++ch) {
    const int t0 = ch * CHUNK;

    if (ch + 1 < NCHUNK) {   // prefetch next chunk into cache hierarchy
      __builtin_prefetch(qg + (size_t)(t0 + CHUNK) * DHD + lane * 8, 0, 1);
      __builtin_prefetch(kg + (size_t)(t0 + CHUNK) * DHD + lane * 8, 0, 1);
      __builtin_prefetch(vg + (size_t)(t0 + CHUNK) * DHD + lane * 8, 0, 1);
    }

    // ---- gates: max-plus inclusive scan over the 32 chunk steps (lane == step) ----
    float fv = fgp[t0 + lane];
    float iv = igp[t0 + lane];
    float a  = fminf(fv, 0.f) - log1pf(__expf(-fabsf(fv)));   // logsigmoid
    float A = a, M = iv;            // (A,M): m_out = max(m_in + A, M), associative
    #pragma unroll
    for (int off = 1; off < 32; off <<= 1) {
      float Ap = __shfl_up(A, off);
      float Mp = __shfl_up(M, off);
      if (lane >= off) { M = fmaxf(Mp + A, M); A = Ap + A; }
    }
    float b      = A;                              // cumsum of logsigmoid(f)
    float mt     = fmaxf(m_prev + b, M);           // exact reference stabilizer
    float g      = __expf(b + m_prev - mt);        // <= 1
    float rowfac = b - mt;
    float colfac = iv - b;
    float em     = __expf(-mt);
    float rl     = __shfl(rowfac, 31);
    float w      = __expf(rl + colfac) * INV_SQRT_DH;  // end-of-chunk weight * 1/sqrt(d)
    float gL     = __shfl(g, 31);
    float m_last = __shfl(mt, 31);
    rowfac_l[lane] = rowfac; colfac_l[lane] = colfac; w_l[lane] = w; g_l[lane] = g;

    // ---- stage Q (f32), K (bf16), V slab (f32), coalesced float4 loads ----
    #pragma unroll
    for (int it = 0; it < 16; ++it) {
      int idx = lane + it * 32;
      int row = idx >> 4, c4 = (idx & 15) << 2;
      float4 qv = *(const float4*)(qg + (size_t)(t0 + row) * DHD + c4);
      *(float2*)&Qc[row][c4]     = make_float2(qv.x, qv.y);
      *(float2*)&Qc[row][c4 + 2] = make_float2(qv.z, qv.w);
      float4 kv = *(const float4*)(kg + (size_t)(t0 + row) * DHD + c4);
      Kb[row][c4+0] = (__bf16)kv.x; Kb[row][c4+1] = (__bf16)kv.y;
      Kb[row][c4+2] = (__bf16)kv.z; Kb[row][c4+3] = (__bf16)kv.w;
    }
    #pragma unroll
    for (int it = 0; it < 4; ++it) {
      int idx = lane + it * 32;
      int row = idx >> 2, c4 = (idx & 3) << 2;
      float4 vv = *(const float4*)(vg + (size_t)(t0 + row) * DHD + c0 + c4);
      *(float2*)&Vc[row][c4]     = make_float2(vv.x, vv.y);
      *(float2*)&Vc[row][c4 + 2] = make_float2(vv.z, vv.w);
    }

    // ---- hoisted per-row gate values (batched LDS reads, no branchy selects) ----
    float rf[2][8], gf[2][8];
    #pragma unroll
    for (int rt = 0; rt < 2; ++rt)
      #pragma unroll
      for (int r = 0; r < 8; ++r) {
        int t = rt * 16 + r + hl * 8;
        rf[rt][r] = rowfac_l[t];
        gf[rt][r] = g_l[t];
      }

    // ---- Q A-fragments (reused by QK^T and Q@C0): 16x32 bf16 A layout ----
    v16bf aQ[2][2];
    #pragma unroll
    for (int rt = 0; rt < 2; ++rt)
      #pragma unroll
      for (int kk = 0; kk < 2; ++kk)
        #pragma unroll
        for (int e = 0; e < 16; ++e) {
          int kcol = kk * 32 + e + hl * 16;
          aQ[rt][kk][e] = (__bf16)Qc[rt * 16 + l15][kcol];
        }

    // ---- K^T B-fragments: built once per column-tile (raw bf16 copies) ----
    v16bf bK[2][2];   // [ct][kk]
    #pragma unroll
    for (int ct = 0; ct < 2; ++ct)
      #pragma unroll
      for (int kk = 0; kk < 2; ++kk)
        #pragma unroll
        for (int e = 0; e < 16; ++e)
          bK[ct][kk][e] = Kb[ct * 16 + l15][kk * 32 + e + hl * 16];

    // ---- S = D ⊙ (Q K^T / sqrt(DH)) for the 3 live tiles ----
    #pragma unroll
    for (int rt = 0; rt < 2; ++rt)
      #pragma unroll
      for (int ct = 0; ct < 2; ++ct) {
        if (rt == 0 && ct == 1) continue;
        v8f accS = {};
        accS = __builtin_amdgcn_wmma_f32_16x16x32_bf16(false, aQ[rt][0], false, bK[ct][0], (short)0, accS, false, false);
        accS = __builtin_amdgcn_wmma_f32_16x16x32_bf16(false, aQ[rt][1], false, bK[ct][1], (short)0, accS, false, false);
        int   scol = ct * 16 + l15;
        float cf   = colfac_l[scol];
        #pragma unroll
        for (int r = 0; r < 8; ++r) {
          int t = rt * 16 + r + hl * 8;
          // branchless: both select operands are registers -> v_cndmask
          float arg = rf[rt][r] + cf;
          arg = (scol <= t) ? arg : -1e30f;
          float val = accS[r] * INV_SQRT_DH * __expf(arg);
          Sl[t][scol] = (__bf16)val;
        }
      }

    // ---- denominator: qn_t = g_t * (q_t . n0) + rowsum_s S[t][s] ----
    float qn_intra = 0.f;
    #pragma unroll 8
    for (int s = 0; s < 32; ++s) qn_intra += (float)Sl[lane][s];
    float qn0 = 0.f;
    #pragma unroll 8
    for (int d = 0; d < 64; ++d) qn0 += Qc[lane][d] * n_lds[d];
    float qn  = g * qn0 + qn_intra;
    rden_l[lane] = 1.f / (fmaxf(fabsf(qn), em) + EPSV);

    // ---- V slab B-fragments: plain (for S@V) and w-scaled (for state) ----
    v16bf bV, bVw;
    #pragma unroll
    for (int e = 0; e < 16; ++e) {
      int s = e + hl * 16;
      float vv = Vc[s][l15];
      bV[e]  = (__bf16)vv;
      bVw[e] = (__bf16)(w_l[s] * vv);
    }

    // ---- H = diag(g) * (Q @ C0) + S @ V ----
    v8f accH[2];
    #pragma unroll
    for (int rt = 0; rt < 2; ++rt) {
      v8f acc = {};
      #pragma unroll
      for (int kk = 0; kk < 2; ++kk) {
        v16bf bC;
        #pragma unroll
        for (int e = 0; e < 16; ++e)
          bC[e] = Cb[kk * 32 + e + hl * 16][l15];
        acc = __builtin_amdgcn_wmma_f32_16x16x32_bf16(false, aQ[rt][kk], false, bC, (short)0, acc, false, false);
      }
      #pragma unroll
      for (int r = 0; r < 8; ++r) acc[r] *= gf[rt][r];
      v16bf aS;
      #pragma unroll
      for (int e = 0; e < 16; ++e)
        aS[e] = Sl[rt * 16 + l15][e + hl * 16];
      acc = __builtin_amdgcn_wmma_f32_16x16x32_bf16(false, aS, false, bV, (short)0, acc, false, false);
      accH[rt] = acc;
    }

    // ---- write h = H * (1/denom); hoist the reciprocal reads ----
    float rd[2][8];
    #pragma unroll
    for (int rt = 0; rt < 2; ++rt)
      #pragma unroll
      for (int r = 0; r < 8; ++r)
        rd[rt][r] = rden_l[rt * 16 + r + hl * 8];
    #pragma unroll
    for (int rt = 0; rt < 2; ++rt)
      #pragma unroll
      for (int r = 0; r < 8; ++r) {
        int t = rt * 16 + r + hl * 8;
        og[(size_t)(t0 + t) * DHD + c0 + l15] = accH[rt][r] * rd[rt][r];
      }

    // ---- state update: C = gL*C + K^T @ (w⊙V)  (f32 WMMA accumulators) ----
    #pragma unroll
    for (int rt = 0; rt < 4; ++rt) {
      v16bf aK;
      #pragma unroll
      for (int e = 0; e < 16; ++e) {
        int s = e + hl * 16;
        aK[e] = Kb[s][rt * 16 + l15];
      }
      v8f acc = accC[rt];
      #pragma unroll
      for (int r = 0; r < 8; ++r) acc[r] *= gL;
      acc = __builtin_amdgcn_wmma_f32_16x16x32_bf16(false, aK, false, bVw, (short)0, acc, false, false);
      accC[rt] = acc;
      #pragma unroll
      for (int r = 0; r < 8; ++r)
        Cb[rt * 16 + r + hl * 8][l15] = (__bf16)acc[r];
    }

    // ---- n update: n = gL*n + sum_s w_s * K_s ----
    #pragma unroll 1
    for (int half = 0; half < 2; ++half) {
      int dk = lane + half * 32;
      float acc = gL * n_lds[dk];
      #pragma unroll 8
      for (int s = 0; s < 32; ++s) acc += w_l[s] * (float)Kb[s][dk];
      n_lds[dk] = acc;
    }

    m_prev = m_last;
  }

  // ---- final states: [h | cT | nT | mT] flat ----
  float* outC = dout + (size_t)BH * S_LEN * DHD;
  #pragma unroll
  for (int rt = 0; rt < 4; ++rt)
    #pragma unroll
    for (int r = 0; r < 8; ++r) {
      int dk = rt * 16 + r + hl * 8;
      outC[(size_t)bh * DHD * DHD + (size_t)dk * DHD + c0 + l15] = accC[rt][r];
    }
  if (slab == 0) {
    float* outN = outC + (size_t)BH * DHD * DHD;
    outN[bh * DHD + lane]      = n_lds[lane];
    outN[bh * DHD + lane + 32] = n_lds[lane + 32];
    if (lane == 0) (outN + BH * DHD)[bh] = m_prev;
  }
}

extern "C" void kernel_launch(void* const* d_in, const int* in_sizes, int n_in,
                              void* d_out, int out_size, void* d_ws, size_t ws_size,
                              hipStream_t stream)
{
  (void)in_sizes; (void)n_in; (void)out_size; (void)d_ws; (void)ws_size;
  mlstm_chunk_wmma<<<dim3(BH * SLABS), dim3(32), 0, stream>>>(
      (const float*)d_in[0],  // q
      (const float*)d_in[1],  // k
      (const float*)d_in[2],  // v
      (const float*)d_in[3],  // i
      (const float*)d_in[4],  // f
      (const float*)d_in[5],  // c_state
      (const float*)d_in[6],  // n_state
      (const float*)d_in[7],  // m_state
      (float*)d_out);
}
